// Block_17446157156413
// MI455X (gfx1250) — matrix operand order repaired and checked by reading, not compiled, not run on hardware
//
#include <hip/hip_runtime.h>
#include <hip/hip_bf16.h>
#include <math.h>

#define Bsz 2048
#define Tt  64
#define Cc  128
#define Hh  8
#define HDd 16
#define FFd 512   // 4*C

typedef __attribute__((ext_vector_type(16))) _Float16 v16h;
typedef __attribute__((ext_vector_type(8)))  float    v8f;
typedef __attribute__((ext_vector_type(4)))  unsigned int v4u;
typedef __attribute__((ext_vector_type(8)))  int      v8i;
typedef __attribute__((ext_vector_type(4)))  int      v4i;

__device__ __forceinline__ v8f wmma(v16h a, v16h b, v8f c) {
  return __builtin_amdgcn_wmma_f32_16x16x32_f16(
      /*neg_a=*/false, a, /*neg_b=*/false, b,
      /*c_mod=*/(short)0, c, /*reuse_a=*/false, /*reuse_b=*/false);
}

// ---- TDM: 2D tile (Tt x Cc, f32) between global and LDS --------------------
// D# per CDNA5 ISA ch.8: group0 = {count/type/lds_addr/global_addr},
// group1 = {data_size, tensor_dim0/1, tile_dim0/1, dim0 stride}.
__device__ __forceinline__ void tdm_tile_f32(unsigned long long gaddr,
                                             unsigned lds_off, bool store) {
  v4u g0;
  g0[0] = 1u;                                     // count=1, user descriptor
  g0[1] = lds_off;                                // lds_addr (bytes)
  g0[2] = (unsigned)(gaddr & 0xFFFFFFFFull);      // global_addr[31:0]
  g0[3] = (unsigned)((gaddr >> 32) & 0x01FFFFFFull) | (2u << 30); // addr[56:32], type=2
  v8i g1 = {};
  g1[0] = (int)(2u << 16);                        // data_size = 2 (4 bytes)
  g1[1] = (int)((unsigned)Cc << 16);              // tensor_dim0[15:0] @ bit48
  g1[2] = (int)((unsigned)Tt << 16);              // tensor_dim1[15:0] @ bit80
  g1[3] = (int)((unsigned)Cc << 16);              // tile_dim0 @ bit112
  g1[4] = Tt;                                     // tile_dim1 @ bit128
  g1[5] = Cc;                                     // tensor_dim0_stride @ bit160
  v4i gz = {};
#if __clang_major__ >= 23
  v8i gz8 = {};
  if (store) __builtin_amdgcn_tensor_store_from_lds(g0, g1, gz, gz, gz8, 0);
  else       __builtin_amdgcn_tensor_load_to_lds (g0, g1, gz, gz, gz8, 0);
#else
  if (store) __builtin_amdgcn_tensor_store_from_lds(g0, g1, gz, gz, 0);
  else       __builtin_amdgcn_tensor_load_to_lds (g0, g1, gz, gz, 0);
#endif
}

// ---- fragment loaders (wave32), contiguous K order shared by A and B -----------------
// A 16x32 f16: lanes 0-15 -> M=lane, K=0..15; lanes 16-31 -> M=lane-16,
// K=16..31; VGPR i holds K=kb+2i,kb+2i+1 (contiguous, matching B loaders)
__device__ __forceinline__ v16h load_a16(const _Float16* src, int ld, int m0, int k0) {
  const int lane = threadIdx.x & 31;
  const _Float16* p = src + (m0 + (lane & 15)) * ld + k0 + ((lane >> 4) << 4);
  v16h a;
#pragma unroll
  for (int i = 0; i < 4; ++i) {
    a[2*i]   = p[2*i];
    a[2*i+1] = p[2*i+1];
    a[8+2*i] = p[8+2*i];
    a[9+2*i] = p[8+2*i+1];
  }
  return a;
}

// A with K zero-padding (used for q with HD=16 padded to K=32)
__device__ __forceinline__ v16h load_a16_pad(const _Float16* src, int ld, int m0,
                                             int k0, int kmax) {
  const int lane = threadIdx.x & 31;
  const int kb = k0 + ((lane >> 4) << 4);
  const _Float16* row = src + (m0 + (lane & 15)) * ld;
  v16h a;
#pragma unroll
  for (int i = 0; i < 4; ++i) {
    int k1 = kb + 2*i;
    int k2 = kb + 8 + 2*i;
    a[2*i]   = (k1     < kmax) ? row[k1]     : (_Float16)0.f;
    a[2*i+1] = (k1 + 1 < kmax) ? row[k1 + 1] : (_Float16)0.f;
    a[8+2*i] = (k2     < kmax) ? row[k2]     : (_Float16)0.f;
    a[9+2*i] = (k2 + 1 < kmax) ? row[k2 + 1] : (_Float16)0.f;
  }
  return a;
}

// B 32x16 f16 from K-major (transposed) storage: element (K,N) = src[n*ld + K].
// lanes 0-15: N=lane, K=0..15; lanes 16-31: N=lane-16, K=16..31 (2 per VGPR).
__device__ __forceinline__ v16h load_bT(const _Float16* src, int ld, int k0, int n0) {
  const int lane = threadIdx.x & 31;
  const _Float16* p = src + (n0 + (lane & 15)) * ld + k0 + ((lane >> 4) << 4);
  v16h b;
#pragma unroll
  for (int i = 0; i < 8; ++i) { b[2*i] = p[2*i]; b[2*i+1] = p[2*i+1]; }
  return b;
}

__device__ __forceinline__ v16h load_bT_pad(const _Float16* src, int ld, int k0,
                                            int n0, int kmax) {
  const int lane = threadIdx.x & 31;
  const int kb = k0 + ((lane >> 4) << 4);
  const _Float16* col = src + (n0 + (lane & 15)) * ld;
  v16h b;
#pragma unroll
  for (int i = 0; i < 8; ++i) {
    int k = kb + 2*i;
    b[2*i]   = (k     < kmax) ? col[k]     : (_Float16)0.f;
    b[2*i+1] = (k + 1 < kmax) ? col[k + 1] : (_Float16)0.f;
  }
  return b;
}

// B 32x16 f16 from row-major storage: element (K,N) = src[(k0+K)*ld + n0+n]
__device__ __forceinline__ v16h load_bRM(const _Float16* src, int ld, int k0, int n0) {
  const int lane = threadIdx.x & 31;
  const _Float16* p = src + n0 + (lane & 15);
  const int kb = k0 + ((lane >> 4) << 4);
  v16h b;
#pragma unroll
  for (int i = 0; i < 8; ++i) {
    b[2*i]   = p[(kb + 2*i)     * ld];
    b[2*i+1] = p[(kb + 2*i + 1) * ld];
  }
  return b;
}

// D/C 16x16 f32: lanes 0-15: N=lane, VGPR p -> M=p; lanes 16-31: N=lane-16, M=8+p
__device__ __forceinline__ void store_d_f16(_Float16* dst, int ld, int m0, int n0, v8f d) {
  const int lane = threadIdx.x & 31;
  const int n  = n0 + (lane & 15);
  const int mh = m0 + ((lane >> 4) << 3);
#pragma unroll
  for (int p = 0; p < 8; ++p) dst[(mh + p) * ld + n] = (_Float16)d[p];
}

// ---- LayerNorm over 128-wide rows; each wave handles 8 rows ----------------
__device__ __forceinline__ void layernorm_rows(const float* __restrict__ src,
                                               _Float16* __restrict__ dst,
                                               const float* __restrict__ gamma,
                                               const float* __restrict__ beta,
                                               int wave, int lane) {
#pragma unroll
  for (int i = 0; i < 8; ++i) {
    const int r = wave * 8 + i;
    const float* row = src + r * Cc;
    float s = 0.f, ss = 0.f;
#pragma unroll
    for (int j = 0; j < 4; ++j) { float v = row[lane + 32*j]; s += v; ss += v*v; }
#pragma unroll
    for (int o = 16; o > 0; o >>= 1) { s += __shfl_xor(s, o); ss += __shfl_xor(ss, o); }
    const float mean = s * (1.f / Cc);
    const float var  = ss * (1.f / Cc) - mean * mean;
    const float rstd = rsqrtf(var + 1e-5f);
#pragma unroll
    for (int j = 0; j < 4; ++j) {
      const int c = lane + 32*j;
      dst[r * Cc + c] = (_Float16)((row[c] - mean) * rstd * gamma[c] + beta[c]);
    }
  }
}

// ---- fp32 -> f16 transpose-convert (per batch slice): dst[c*rows+r]=src[r*cols+c]
__global__ void cvt_t_batched(const float* __restrict__ src, _Float16* __restrict__ dst,
                              int rows, int cols) {
  const int n = rows * cols;
  src += (size_t)blockIdx.y * n;
  dst += (size_t)blockIdx.y * n;
  for (int i = blockIdx.x * blockDim.x + threadIdx.x; i < n; i += gridDim.x * blockDim.x) {
    const int r = i / cols, c = i % cols;
    dst[c * rows + r] = (_Float16)src[i];
  }
}

// ---- fused transformer block: one workgroup (8 waves) per batch element ----
// LDS budget: 160 KB -> 2 workgroups per 320 KB WGP.
__global__ __launch_bounds__(256)
void block_fwd(const float* __restrict__ x,
               const _Float16* __restrict__ wqh, const _Float16* __restrict__ wkh,
               const _Float16* __restrict__ wvh, const _Float16* __restrict__ wph,
               const _Float16* __restrict__ w1h, const _Float16* __restrict__ w2h,
               const float* __restrict__ bproj, const float* __restrict__ b1,
               const float* __restrict__ b2,
               const float* __restrict__ g1, const float* __restrict__ be1,
               const float* __restrict__ g2, const float* __restrict__ be2,
               float* __restrict__ out)
{
  extern __shared__ char smem[];
  float*    xs  = (float*)(smem);                //   0 KB: 64x128 f32 residual (32 KB)
  _Float16* hs  = (_Float16*)(smem + 32768);     //  32 KB: 64x128 f16 LN out   (16 KB)
  _Float16* qs  = (_Float16*)(smem + 49152);     //  48 KB: [8][64][16] f16     (16 KB)
  _Float16* ac  = (_Float16*)(smem + 49152);     //  union: 64x128 attn concat (after q dead)
  _Float16* ks  = (_Float16*)(smem + 65536);     //  64 KB
  _Float16* vs  = (_Float16*)(smem + 81920);     //  80 KB
  _Float16* scb = (_Float16*)(smem + 98304);     //  96 KB: 8 x (64x64 f16)    (64 KB)
  _Float16* ff  = (_Float16*)(smem + 98304);     //  union: 64x512 f16 FF buf  (64 KB)

  const int tid  = threadIdx.x;
  const int lane = tid & 31;
  const int wid  = tid >> 5;     // wave 0..7 == head id
  const int b    = blockIdx.x;

  // phase 0: TDM-stage x[b] (64x128 f32 tile) into LDS; wave 0 drives the DMA
  if (wid == 0) {
    tdm_tile_f32((unsigned long long)(const void*)(x + (size_t)b * Tt * Cc),
                 /*lds_off=*/0u, /*store=*/false);
    __builtin_amdgcn_s_wait_tensorcnt((short)0);
  }
  __syncthreads();

  // phase 1: LN1 -> hs (f16)
  layernorm_rows(xs, hs, g1, be1, wid, lane);
  __syncthreads();

  // phase 2: QKV for head `wid`  (h[64x128] x w[128x16], K tiles of 32)
  {
    const _Float16* wq_h = wqh + wid * Cc * HDd;   // K-major [16][128]
    const _Float16* wk_h = wkh + wid * Cc * HDd;
    const _Float16* wv_h = wvh + wid * Cc * HDd;
    _Float16* qh = qs + wid * Tt * HDd;
    _Float16* kh = ks + wid * Tt * HDd;
    _Float16* vh = vs + wid * Tt * HDd;
#pragma unroll
    for (int mt = 0; mt < 4; ++mt) {
      v8f cq = {}, ck = {}, cv = {};
#pragma unroll
      for (int kt = 0; kt < 4; ++kt) {
        v16h a = load_a16(hs, Cc, mt * 16, kt * 32);
        cq = wmma(a, load_bT(wq_h, Cc, kt * 32, 0), cq);
        ck = wmma(a, load_bT(wk_h, Cc, kt * 32, 0), ck);
        cv = wmma(a, load_bT(wv_h, Cc, kt * 32, 0), cv);
      }
      store_d_f16(qh, HDd, mt * 16, 0, cq);
      store_d_f16(kh, HDd, mt * 16, 0, ck);
      store_d_f16(vh, HDd, mt * 16, 0, cv);
    }

    // phase 3a: scores = q @ k^T for head `wid` (K=16 zero-padded to 32)
    _Float16* sc = scb + wid * (Tt * Tt);          // wave-private 64x64 f16
#pragma unroll
    for (int mt = 0; mt < 4; ++mt) {
      v16h a = load_a16_pad(qh, HDd, mt * 16, 0, HDd);
#pragma unroll
      for (int nt = 0; nt < 4; ++nt) {
        v8f c = {};
        c = wmma(a, load_bT_pad(kh, HDd, 0, nt * 16, HDd), c);
        store_d_f16(sc, Tt, mt * 16, nt * 16, c);
      }
    }
    // phase 3b: causal softmax (2 cols/lane), scale = HD^-0.5 = 0.25
    for (int r = 0; r < Tt; ++r) {
      const int c1 = lane + 32;
      float v0 = (lane <= r) ? (float)sc[r * Tt + lane] * 0.25f : -3.0e38f;
      float v1 = (c1   <= r) ? (float)sc[r * Tt + c1]   * 0.25f : -3.0e38f;
      float m = fmaxf(v0, v1);
#pragma unroll
      for (int o = 16; o > 0; o >>= 1) m = fmaxf(m, __shfl_xor(m, o));
      float e0 = (lane <= r) ? __expf(v0 - m) : 0.f;
      float e1 = (c1   <= r) ? __expf(v1 - m) : 0.f;
      float s = e0 + e1;
#pragma unroll
      for (int o = 16; o > 0; o >>= 1) s += __shfl_xor(s, o);
      const float inv = 1.f / s;
      sc[r * Tt + lane] = (_Float16)(e0 * inv);
      sc[r * Tt + c1]   = (_Float16)(e1 * inv);
    }
    // all waves done reading q -> `ac` may now overlay `qs`
    __syncthreads();

    // phase 3c: attn = P @ V -> columns [wid*16, wid*16+16) of concat buffer
    const _Float16* vh2 = vs + wid * Tt * HDd;
#pragma unroll
    for (int mt = 0; mt < 4; ++mt) {
      v8f c = {};
#pragma unroll
      for (int kt = 0; kt < 2; ++kt)
        c = wmma(load_a16(sc, Tt, mt * 16, kt * 32),
                 load_bRM(vh2, HDd, kt * 32, 0), c);
      store_d_f16(ac, Cc, mt * 16, wid * HDd, c);
    }
  }
  __syncthreads();

  // phase 4: proj + bias + residual into xs (32 tiles over 8 waves)
  for (int t = wid * 4; t < wid * 4 + 4; ++t) {
    const int mt = t >> 3, nt = t & 7;
    v8f c = {};
#pragma unroll
    for (int kt = 0; kt < 4; ++kt)
      c = wmma(load_a16(ac, Cc, mt * 16, kt * 32),
               load_bT(wph, Cc, kt * 32, nt * 16), c);
    const int n  = nt * 16 + (lane & 15);
    const int m0 = mt * 16 + ((lane >> 4) << 3);
    const float bias = bproj[n];
#pragma unroll
    for (int p = 0; p < 8; ++p) xs[(m0 + p) * Cc + n] += c[p] + bias;
  }
  __syncthreads();

  // phase 5: LN2 -> hs
  layernorm_rows(xs, hs, g2, be2, wid, lane);
  __syncthreads();

  // phase 6: FF1 + GELU(exact) -> ff (128 tiles over 8 waves)
  for (int t = wid * 16; t < wid * 16 + 16; ++t) {
    const int mt = t >> 5, nt = t & 31;
    v8f c = {};
#pragma unroll
    for (int kt = 0; kt < 4; ++kt)
      c = wmma(load_a16(hs, Cc, mt * 16, kt * 32),
               load_bT(w1h, Cc, kt * 32, nt * 16), c);
    const int n  = nt * 16 + (lane & 15);
    const int m0 = mt * 16 + ((lane >> 4) << 3);
    const float bias = b1[n];
#pragma unroll
    for (int p = 0; p < 8; ++p) {
      const float v = c[p] + bias;
      const float g = 0.5f * v * (1.f + erff(v * 0.70710678118f));
      ff[(m0 + p) * FFd + n] = (_Float16)g;
    }
  }
  __syncthreads();

  // phase 7: FF2 + bias + residual -> xs (in place; same lane owns read+write)
  for (int t = wid * 4; t < wid * 4 + 4; ++t) {
    const int mt = t >> 3, nt = t & 7;
    v8f c = {};
#pragma unroll
    for (int kt = 0; kt < 16; ++kt)
      c = wmma(load_a16(ff, FFd, mt * 16, kt * 32),
               load_bT(w2h, FFd, kt * 32, nt * 16), c);
    const int n  = nt * 16 + (lane & 15);
    const int m0 = mt * 16 + ((lane >> 4) << 3);
    const float bias = b2[n];
#pragma unroll
    for (int p = 0; p < 8; ++p)
      xs[(m0 + p) * Cc + n] += c[p] + bias;
  }
  __syncthreads();

  // phase 8: TDM-store the finished 64x128 f32 tile to out[b]
  if (wid == 0) {
    tdm_tile_f32((unsigned long long)(void*)(out + (size_t)b * Tt * Cc),
                 /*lds_off=*/0u, /*store=*/true);
    __builtin_amdgcn_s_wait_tensorcnt((short)0);
  }
}

extern "C" void kernel_launch(void* const* d_in, const int* in_sizes, int n_in,
                              void* d_out, int out_size, void* d_ws, size_t ws_size,
                              hipStream_t stream) {
  const float* x     = (const float*)d_in[0];
  const float* wq    = (const float*)d_in[1];
  const float* wk    = (const float*)d_in[2];
  const float* wv    = (const float*)d_in[3];
  const float* wp    = (const float*)d_in[4];
  const float* bproj = (const float*)d_in[5];
  const float* w1    = (const float*)d_in[6];
  const float* b1    = (const float*)d_in[7];
  const float* w2    = (const float*)d_in[8];
  const float* b2    = (const float*)d_in[9];
  const float* g1    = (const float*)d_in[10];
  const float* be1   = (const float*)d_in[11];
  const float* g2    = (const float*)d_in[12];
  const float* be2   = (const float*)d_in[13];
  float* out = (float*)d_out;

  // f16 transposed weights in workspace (K-major per output column)
  _Float16* ws  = (_Float16*)d_ws;
  _Float16* wqh = ws;              // 8*128*16 = 16384
  _Float16* wkh = ws + 16384;
  _Float16* wvh = ws + 32768;
  _Float16* wph = ws + 49152;      // 128*128  = 16384
  _Float16* w1h = ws + 65536;      // 128*512  = 65536
  _Float16* w2h = ws + 131072;     // 512*128  = 65536

  cvt_t_batched<<<dim3(8, 8),   256, 0, stream>>>(wq, wqh, 128, 16);
  cvt_t_batched<<<dim3(8, 8),   256, 0, stream>>>(wk, wkh, 128, 16);
  cvt_t_batched<<<dim3(8, 8),   256, 0, stream>>>(wv, wvh, 128, 16);
  cvt_t_batched<<<dim3(64, 1),  256, 0, stream>>>(wp, wph, 128, 128);
  cvt_t_batched<<<dim3(256, 1), 256, 0, stream>>>(w1, w1h, 128, 512);
  cvt_t_batched<<<dim3(256, 1), 256, 0, stream>>>(w2, w2h, 512, 128);

  const size_t smem_bytes = 98304 + 65536;   // 160 KB -> 2 workgroups/WGP
  hipFuncSetAttribute((const void*)block_fwd,
                      hipFuncAttributeMaxDynamicSharedMemorySize, (int)smem_bytes);
  block_fwd<<<Bsz, 256, smem_bytes, stream>>>(x, wqh, wkh, wvh, wph, w1h, w2h,
                                              bproj, b1, b2, g1, be1, g2, be2, out);
}
